// PropAndPool_54631984005432
// MI455X (gfx1250) — matrix operand order repaired and checked by reading, not compiled, not run on hardware
//
#include <hip/hip_runtime.h>
#include <hip/hip_bf16.h>
#include <math.h>

// Problem constants (fixed by the reference's setup_inputs)
#define BG    64
#define NPG0  1024
#define DM    128
#define DEG   16
#define E_TOT (BG * NPG0 * DEG)   // 1,048,576 edges
#define N0    (BG * NPG0)         // 65,536 nodes
#define WSTR  136                 // LDS weight row stride in halves (272B: 16B-aligned, bank-spread)

typedef __attribute__((ext_vector_type(16))) _Float16 v16h;
typedef __attribute__((ext_vector_type(8)))  _Float16 v8h;
typedef __attribute__((ext_vector_type(8)))  float    v8f;

__device__ __forceinline__ void atomAddF(float* p, float v) {
  (void)__hip_atomic_fetch_add(p, v, __ATOMIC_RELAXED, __HIP_MEMORY_SCOPE_AGENT);
}

// ---------------------------------------------------------------------------
// Init: copy edges into workspace (mutable), mask=1, zero readout accumulator
// ---------------------------------------------------------------------------
__global__ void init_edges_kernel(const int* __restrict__ ei,
                                  int* __restrict__ srcW, int* __restrict__ dstW,
                                  float* __restrict__ maskW, float* __restrict__ sacc) {
  int t = blockIdx.x * blockDim.x + threadIdx.x;
  if (t < E_TOT) {
    srcW[t]  = ei[t];
    dstW[t]  = ei[E_TOT + t];
    maskW[t] = 1.0f;
  }
  if (t < BG * 256) sacc[t] = 0.0f;
}

__global__ void zero_agg_kernel(float* __restrict__ agg, float* __restrict__ cnt, int n) {
  int t = blockIdx.x * blockDim.x + threadIdx.x;
  if (t < n * DM) agg[t] = 0.0f;
  if (t < n)      cnt[t] = 0.0f;
}

// ---------------------------------------------------------------------------
// Edge scatter: agg[dst] += x[src]*mask ; cnt[dst] += mask
// One thread handles 32 features (8 float4) of one edge.
// ---------------------------------------------------------------------------
__global__ void scatter_kernel(const float* __restrict__ x,
                               const int* __restrict__ src, const int* __restrict__ dst,
                               const float* __restrict__ mask,
                               float* __restrict__ agg, float* __restrict__ cnt) {
  int t = blockIdx.x * blockDim.x + threadIdx.x;
  int e = t >> 2;
  if (e >= E_TOT) return;
  float m = mask[e];
  if (m == 0.0f) return;
  int c = (t & 3) * 32;
  int s = src[e], d = dst[e];
  const float4* xs = (const float4*)(x + (size_t)s * DM + c);
  float* ag = agg + (size_t)d * DM + c;
#pragma unroll
  for (int i = 0; i < 8; ++i) {
    float4 v = xs[i];
    atomAddF(ag + 4 * i + 0, v.x * m);
    atomAddF(ag + 4 * i + 1, v.y * m);
    atomAddF(ag + 4 * i + 2, v.z * m);
    atomAddF(ag + 4 * i + 3, v.w * m);
  }
  if ((t & 3) == 0) atomAddF(cnt + d, m);
}

// ---------------------------------------------------------------------------
// SAGE linear: h = relu( (agg/max(cnt,1)) @ Wl + bl + x @ Wr )
// One wave32 per 16-node tile; V_WMMA_F32_16X16X32_F16, f32 accumulate.
// Weights are staged once per block into LDS, f16, transposed to [n][k] with a
// padded stride so B fragments are two 16B ds loads per matrix (no rescan of
// global memory inside the nt/kt loops).
// A layout (16-bit 16x32, this code's contiguous k order): lane M = lane&15, half = lane>>4,
//   elem i -> K = half*16 + i
// B layout (32x16): lane N = lane&15, elem e -> K = half*16 + e
// C/D layout: lane L, reg r -> M = r + 8*(L>>4), N = L&15
// ---------------------------------------------------------------------------
__global__ __launch_bounds__(256) void sage_wmma_kernel(
    const float* __restrict__ xcur, const float* __restrict__ agg,
    const float* __restrict__ cnt,
    const float* __restrict__ Wl, const float* __restrict__ bl,
    const float* __restrict__ Wr,
    float* __restrict__ h, int ntiles) {
  __shared__ _Float16 sWl[DM * WSTR];
  __shared__ _Float16 sWr[DM * WSTR];

  int tid = threadIdx.x;
  // Cooperative stage: sW[n*WSTR + k] = (f16) W[k*128 + n]   (coalesced global reads)
  for (int i = tid; i < DM * DM; i += 256) {
    int k = i >> 7, n = i & 127;
    sWl[n * WSTR + k] = (_Float16)Wl[i];
    sWr[n * WSTR + k] = (_Float16)Wr[i];
  }
  __syncthreads();

  int wave = (blockIdx.x * blockDim.x + tid) >> 5;
  int lane = tid & 31;
  if (wave >= ntiles) return;               // wave-uniform, after the barrier

  int m16  = lane & 15;
  int hh   = lane >> 4;
  int node = wave * 16 + m16;
  float inv = 1.0f / fmaxf(cnt[node], 1.0f);
  const float* arow = agg  + (size_t)node * DM;
  const float* xrow = xcur + (size_t)node * DM;

  v16h aM[4], aX[4];
#pragma unroll
  for (int kt = 0; kt < 4; ++kt) {
#pragma unroll
    for (int i = 0; i < 16; ++i) {
      int k = kt * 32 + hh * 16 + i;
      aM[kt][i] = (_Float16)(arow[k] * inv);
      aX[kt][i] = (_Float16)(xrow[k]);
    }
  }

  for (int nt = 0; nt < 8; ++nt) {
    int ncol = nt * 16 + m16;
    float bv = bl[ncol];
    v8f acc;
#pragma unroll
    for (int r = 0; r < 8; ++r) acc[r] = bv;

    const _Float16* rowL = &sWl[ncol * WSTR + hh * 16];
    const _Float16* rowR = &sWr[ncol * WSTR + hh * 16];
#pragma unroll
    for (int kt = 0; kt < 4; ++kt) {
      const v8h* pl = (const v8h*)(rowL + kt * 32);   // 16B-aligned
      const v8h* pr = (const v8h*)(rowR + kt * 32);
      v8h l0 = pl[0], l1 = pl[1];
      v8h r0 = pr[0], r1 = pr[1];
      v16h bM, bX;
#pragma unroll
      for (int e = 0; e < 8; ++e) {
        bM[e] = l0[e]; bM[8 + e] = l1[e];
        bX[e] = r0[e]; bX[8 + e] = r1[e];
      }
      acc = __builtin_amdgcn_wmma_f32_16x16x32_f16(false, aM[kt], false, bM,
                                                   (short)0, acc, false, false);
      acc = __builtin_amdgcn_wmma_f32_16x16x32_f16(false, aX[kt], false, bX,
                                                   (short)0, acc, false, false);
    }
#pragma unroll
    for (int r = 0; r < 8; ++r) {
      int mrow = r + 8 * hh;
      h[(size_t)(wave * 16 + mrow) * DM + ncol] = fmaxf(acc[r], 0.0f);
    }
  }
}

// ---------------------------------------------------------------------------
// TopK pool per graph: score = tanh(h.pw/||pw||); bitonic sort (desc) in LDS;
// write x_new = h[perm]*score, build newid (old -> new or -1).
// ---------------------------------------------------------------------------
__global__ __launch_bounds__(1024) void topk_kernel(
    const float* __restrict__ h, const float* __restrict__ pw,
    float* __restrict__ xnew, int* __restrict__ newid, int Npg, int K) {
  __shared__ float skey[1024];
  __shared__ int   sidx[1024];
  __shared__ float sinv;
  int g = blockIdx.x, t = threadIdx.x;

  if (t == 0) {
    float s = 0.0f;
    for (int k = 0; k < DM; ++k) s += pw[k] * pw[k];
    sinv = rsqrtf(s);
  }
  __syncthreads();

  float key = -INFINITY;
  if (t < Npg) {
    const float* row = h + (size_t)(g * Npg + t) * DM;
    float dv = 0.0f;
    for (int k = 0; k < DM; ++k) dv += row[k] * pw[k];
    key = tanhf(dv * sinv);
  }
  skey[t] = key;
  sidx[t] = t;
  __syncthreads();

  // bitonic sort, descending
  for (int k = 2; k <= 1024; k <<= 1) {
    for (int j = k >> 1; j > 0; j >>= 1) {
      int ixj = t ^ j;
      if (ixj > t) {
        bool desc = ((t & k) == 0);
        float k1 = skey[t], k2 = skey[ixj];
        bool sw = desc ? (k1 < k2) : (k1 > k2);
        if (sw) {
          skey[t] = k2; skey[ixj] = k1;
          int tmp = sidx[t]; sidx[t] = sidx[ixj]; sidx[ixj] = tmp;
        }
      }
      __syncthreads();
    }
  }

  if (t < Npg) newid[g * Npg + t] = -1;
  __syncthreads();
  if (t < K) newid[g * Npg + sidx[t]] = g * K + t;

  for (int i = t; i < K * DM; i += 1024) {
    int r = i >> 7, f = i & (DM - 1);
    xnew[(size_t)(g * K + r) * DM + f] =
        h[(size_t)(g * Npg + sidx[r]) * DM + f] * skey[r];
  }
}

// ---------------------------------------------------------------------------
// Edge remap after pooling (in place, elementwise)
// ---------------------------------------------------------------------------
__global__ void remap_edges_kernel(int* __restrict__ src, int* __restrict__ dst,
                                   float* __restrict__ mask, const int* __restrict__ newid) {
  int e = blockIdx.x * blockDim.x + threadIdx.x;
  if (e >= E_TOT) return;
  float m = mask[e];
  int s2 = newid[src[e]];
  int d2 = newid[dst[e]];
  mask[e] = (s2 >= 0 && d2 >= 0) ? m : 0.0f;
  src[e]  = s2 > 0 ? s2 : 0;
  dst[e]  = d2 > 0 ? d2 : 0;
}

// ---------------------------------------------------------------------------
// Readout: sacc[g] += concat(max over K, mean over K)
// ---------------------------------------------------------------------------
__global__ __launch_bounds__(128) void readout_kernel(const float* __restrict__ x,
                                                      float* __restrict__ sacc, int K) {
  int g = blockIdx.x, f = threadIdx.x;
  const float* base = x + (size_t)g * K * DM + f;
  float mx = -INFINITY, sum = 0.0f;
  for (int r = 0; r < K; ++r) {
    float v = base[(size_t)r * DM];
    mx = fmaxf(mx, v);
    sum += v;
  }
  sacc[g * 256 + f]       += mx;
  sacc[g * 256 + 128 + f] += sum / (float)K;
}

// ---------------------------------------------------------------------------
// Final MLP: relu(s@W1+b1) -> relu(@W2+b2) -> sigmoid(@W3+b3)
// ---------------------------------------------------------------------------
__global__ __launch_bounds__(128) void mlp_kernel(
    const float* __restrict__ sacc,
    const float* __restrict__ W1, const float* __restrict__ b1,
    const float* __restrict__ W2, const float* __restrict__ b2,
    const float* __restrict__ W3, const float* __restrict__ b3,
    float* __restrict__ out) {
  __shared__ float h1[128];
  __shared__ float h2[64];
  int g = blockIdx.x, t = threadIdx.x;
  const float* srow = sacc + g * 256;

  float a = b1[t];
  for (int k = 0; k < 256; ++k) a += srow[k] * W1[k * 128 + t];
  h1[t] = fmaxf(a, 0.0f);
  __syncthreads();

  if (t < 64) {
    float a2 = b2[t];
    for (int k = 0; k < 128; ++k) a2 += h1[k] * W2[k * 64 + t];
    h2[t] = fmaxf(a2, 0.0f);
  }
  __syncthreads();

  if (t == 0) {
    float a3 = b3[0];
    for (int k = 0; k < 64; ++k) a3 += h2[k] * W3[k];
    out[g] = 1.0f / (1.0f + expf(-a3));
  }
}

// ---------------------------------------------------------------------------
extern "C" void kernel_launch(void* const* d_in, const int* in_sizes, int n_in,
                              void* d_out, int out_size, void* d_ws, size_t ws_size,
                              hipStream_t stream) {
  (void)in_sizes; (void)n_in; (void)out_size; (void)ws_size;

  const float* x  = (const float*)d_in[0];
  const int*   ei = (const int*)d_in[1];
  const float* W1 = (const float*)d_in[14];
  const float* b1 = (const float*)d_in[15];
  const float* W2 = (const float*)d_in[16];
  const float* b2 = (const float*)d_in[17];
  const float* W3 = (const float*)d_in[18];
  const float* b3 = (const float*)d_in[19];

  // Workspace layout (256B-aligned slots)
  char* base = (char*)d_ws;
  size_t off = 0;
  auto carve = [&](size_t bytes) -> char* {
    char* p = base + off;
    off += (bytes + 255) & ~(size_t)255;
    return p;
  };
  float* xcur  = (float*)carve((size_t)N0 * DM * sizeof(float));
  float* agg   = (float*)carve((size_t)N0 * DM * sizeof(float));
  float* hbuf  = (float*)carve((size_t)N0 * DM * sizeof(float));
  float* cnt   = (float*)carve((size_t)N0 * sizeof(float));
  int*   newid = (int*)  carve((size_t)N0 * sizeof(int));
  int*   srcW  = (int*)  carve((size_t)E_TOT * sizeof(int));
  int*   dstW  = (int*)  carve((size_t)E_TOT * sizeof(int));
  float* maskW = (float*)carve((size_t)E_TOT * sizeof(float));
  float* sacc  = (float*)carve((size_t)BG * 256 * sizeof(float));

  hipMemcpyAsync(xcur, x, (size_t)N0 * DM * sizeof(float),
                 hipMemcpyDeviceToDevice, stream);
  init_edges_kernel<<<(E_TOT + 255) / 256, 256, 0, stream>>>(ei, srcW, dstW, maskW, sacc);

  const int nIn[3]  = {N0, BG * 820, BG * 656};  // 65536, 52480, 41984
  const int NpgL[3] = {NPG0, 820, 656};
  const int Kv[3]   = {820, 656, 525};           // ceil(0.8 * Npg)

  for (int L = 0; L < 3; ++L) {
    const float* Wl = (const float*)d_in[2 + 4 * L];
    const float* bl = (const float*)d_in[3 + 4 * L];
    const float* Wr = (const float*)d_in[4 + 4 * L];
    const float* pw = (const float*)d_in[5 + 4 * L];
    int n = nIn[L];

    zero_agg_kernel<<<(n * DM + 255) / 256, 256, 0, stream>>>(agg, cnt, n);
    scatter_kernel<<<(E_TOT * 4 + 255) / 256, 256, 0, stream>>>(xcur, srcW, dstW, maskW,
                                                                agg, cnt);
    int ntiles = n / 16;
    sage_wmma_kernel<<<(ntiles * 32 + 255) / 256, 256, 0, stream>>>(xcur, agg, cnt,
                                                                    Wl, bl, Wr, hbuf, ntiles);
    // pooled features overwrite xcur (dead after the GEMM)
    topk_kernel<<<BG, 1024, 0, stream>>>(hbuf, pw, xcur, newid, NpgL[L], Kv[L]);
    remap_edges_kernel<<<(E_TOT + 255) / 256, 256, 0, stream>>>(srcW, dstW, maskW, newid);
    readout_kernel<<<BG, 128, 0, stream>>>(xcur, sacc, Kv[L]);
  }

  mlp_kernel<<<BG, 128, 0, stream>>>(sacc, W1, b1, W2, b2, W3, b3, (float*)d_out);
}
